// AdvancedMoEBlock_84172769067205
// MI455X (gfx1250) — compile-verified, matrix operand order checked by reading, not run on hardware
//
#include <hip/hip_runtime.h>
#include <hip/hip_bf16.h>
#include <math.h>

#define H_DIM  1024
#define H2_DIM 2048
#define E_NUM  8

// per-expert swizzled-weight sizes (in 32-bit words): elements/2
#define W1_UINTS ((size_t)H_DIM * H2_DIM / 2)   // 1,048,576
#define W2_UINTS ((size_t)H2_DIM * H_DIM / 2)
#define W1_FRAGS (( (H_DIM / 32) * (H2_DIM / 16) ))   // 4096 fragments / expert
#define W2_FRAGS (( (H2_DIM / 32) * (H_DIM / 16) ))   // 4096 fragments / expert

typedef __attribute__((ext_vector_type(16))) __bf16 v16bf;
typedef __attribute__((ext_vector_type(8)))  float  v8f;

union FragBF { v16bf v; unsigned int u[8]; unsigned short s[16]; };

__device__ __forceinline__ unsigned short f2bf(float f) {
  unsigned int u = __float_as_uint(f);
  u += 0x7FFFu + ((u >> 16) & 1u);          // round-to-nearest-even
  return (unsigned short)(u >> 16);
}

__device__ __forceinline__ float gelu_exact(float v) {
  return 0.5f * v * (1.0f + erff(v * 0.70710678118654752f));
}

// A fragment (16x32 bf16) from a row-major bf16 row: two contiguous 16B runs.
// base4 indexes uint4 units: (k0/2 + kbu)/4 ; second run at +8 uints (+2 uint4).
__device__ __forceinline__ v16bf load_a_frag128(const uint4* __restrict__ row4, int base4) {
  FragBF a;
  uint4 lo = row4[base4];
  uint4 hi = row4[base4 + 2];
  a.u[0] = lo.x; a.u[1] = lo.y; a.u[2] = lo.z; a.u[3] = lo.w;
  a.u[4] = hi.x; a.u[5] = hi.y; a.u[6] = hi.z; a.u[7] = hi.w;
  return a.v;
}

// B fragment (32x16 bf16) from pre-swizzled storage: lane's 32B are contiguous.
__device__ __forceinline__ v16bf load_b_frag128(const uint4* __restrict__ p) {
  FragBF b;
  uint4 lo = p[0];
  uint4 hi = p[1];
  b.u[0] = lo.x; b.u[1] = lo.y; b.u[2] = lo.z; b.u[3] = lo.w;
  b.u[4] = hi.x; b.u[5] = hi.y; b.u[6] = hi.z; b.u[7] = hi.w;
  return b.v;
}

// ---------------------------------------------------------------- init
__global__ void init_kernel(float* __restrict__ out, int out_elems,
                            float* __restrict__ counts, int* __restrict__ cursor) {
  int idx = blockIdx.x * blockDim.x + threadIdx.x;
  int stride = gridDim.x * blockDim.x;
  for (int i = idx; i < out_elems; i += stride) out[i] = 0.0f;
  if (idx < E_NUM) { counts[idx] = 0.0f; cursor[idx] = 0; }
}

// ---------------------------------------------------------------- weight pack
// One wave32 per 32x16 fragment. Converts f32 -> bf16 and swizzles into the
// exact WMMA B-operand register layout:
//   lane L: column n = n0 + (L&15), K-half kb = (L>=16)?8:0,
//   u[i] = pack(k0+kb+2i+(i>=4?8:0), +1)  stored at frag*1KB + L*32B.
__global__ void pack_weights_kernel(const float* __restrict__ W1,
                                    const float* __restrict__ W2,
                                    unsigned int* __restrict__ W1s,
                                    unsigned int* __restrict__ W2s) {
  int gw   = (blockIdx.x * blockDim.x + threadIdx.x) >> 5;
  int lane = threadIdx.x & 31;
  const int total = E_NUM * (W1_FRAGS + W2_FRAGS);
  if (gw >= total) return;

  const float* W; unsigned int* Ws; int K, Nn, e, f;
  if (gw < E_NUM * W1_FRAGS) {
    W = W1; Ws = W1s; K = H_DIM; Nn = H2_DIM;
    e = gw / W1_FRAGS; f = gw % W1_FRAGS;
  } else {
    gw -= E_NUM * W1_FRAGS;
    W = W2; Ws = W2s; K = H2_DIM; Nn = H_DIM;
    e = gw / W2_FRAGS; f = gw % W2_FRAGS;
  }
  const int kf = K / 32;                  // fragments along K (contiguous in memory)
  const int n0 = (f / kf) * 16;
  const int k0 = (f % kf) * 32;

  const float* We = W + (size_t)e * K * Nn;
  unsigned int* dst = Ws + (size_t)e * ((size_t)K * Nn / 2) + (size_t)f * 256 + lane * 8;

  const int n  = n0 + (lane & 15);
  const int kb = (lane >> 4) << 3;

  unsigned int u[8];
#pragma unroll
  for (int i = 0; i < 8; ++i) {
    int k = k0 + kb + 2 * i + ((i >= 4) ? 8 : 0);
    unsigned int lo = f2bf(We[(size_t)k * Nn + n]);
    unsigned int hi = f2bf(We[(size_t)(k + 1) * Nn + n]);
    u[i] = lo | (hi << 16);
  }
  uint4* d4 = (uint4*)dst;
  d4[0] = make_uint4(u[0], u[1], u[2], u[3]);
  d4[1] = make_uint4(u[4], u[5], u[6], u[7]);
}

// ---------------------------------------------------------------- router (wave32 per token)
__global__ void router_kernel(const float* __restrict__ x, const float* __restrict__ Wr,
                              float* __restrict__ gates, float* __restrict__ counts,
                              int* __restrict__ cursor, int* __restrict__ lists,
                              unsigned short* __restrict__ xbf, int N) {
  const int wave = threadIdx.x >> 5;
  const int lane = threadIdx.x & 31;
  const int token = blockIdx.x * (blockDim.x >> 5) + wave;
  if (token >= N) return;

  const float* xr = x + (size_t)token * H_DIM;
  float p[E_NUM];
#pragma unroll
  for (int e = 0; e < E_NUM; ++e) p[e] = 0.0f;

  for (int h = lane; h < H_DIM; h += 32) {
    float xv = xr[h];
    xbf[(size_t)token * H_DIM + h] = f2bf(xv);     // fused bf16 copy of x
#pragma unroll
    for (int e = 0; e < E_NUM; ++e) p[e] += xv * Wr[h * E_NUM + e];
  }
#pragma unroll
  for (int off = 16; off >= 1; off >>= 1) {
#pragma unroll
    for (int e = 0; e < E_NUM; ++e) p[e] += __shfl_xor(p[e], off, 32);
  }

  if (lane == 0) {
    float mx = p[0];
#pragma unroll
    for (int e = 1; e < E_NUM; ++e) mx = fmaxf(mx, p[e]);
    float pr[E_NUM], s = 0.0f;
#pragma unroll
    for (int e = 0; e < E_NUM; ++e) { pr[e] = expf(p[e] - mx); s += pr[e]; }
#pragma unroll
    for (int e = 0; e < E_NUM; ++e) pr[e] /= s;

    int i1 = 0;
#pragma unroll
    for (int e = 1; e < E_NUM; ++e) if (pr[e] > pr[i1]) i1 = e;
    int i2 = (i1 == 0) ? 1 : 0;
#pragma unroll
    for (int e = 0; e < E_NUM; ++e) if (e != i1 && pr[e] > pr[i2]) i2 = e;

    float den = pr[i1] + pr[i2];
    float g1 = pr[i1] / den, g2 = pr[i2] / den;
#pragma unroll
    for (int e = 0; e < E_NUM; ++e) gates[(size_t)token * E_NUM + e] = 0.0f;
    gates[(size_t)token * E_NUM + i1] = g1;
    gates[(size_t)token * E_NUM + i2] = g2;

    atomicAdd(&counts[i1], 1.0f);
    atomicAdd(&counts[i2], 1.0f);
    int p1 = atomicAdd(&cursor[i1], 1); lists[i1 * N + p1] = token;
    int p2 = atomicAdd(&cursor[i2], 1); lists[i2 * N + p2] = token;
  }
}

// ---------------------------------------------------------------- aux loss
__global__ void aux_kernel(const float* __restrict__ counts, float* __restrict__ out,
                           int aux_idx) {
  if (threadIdx.x == 0 && blockIdx.x == 0) {
    float tot = 0.0f;
#pragma unroll
    for (int e = 0; e < E_NUM; ++e) tot += counts[e];
    float aux = 0.0f;
#pragma unroll
    for (int e = 0; e < E_NUM; ++e) {
      float l = counts[e] / tot;
      aux += l * logf(l + 1e-9f);
    }
    out[aux_idx] = 0.01f * aux;
  }
}

// ---------------------------------------------------------------- expert FFN (WMMA bf16)
// block: 256 threads (8 wave32s), 16 gathered tokens, full 1024 out columns.
// hidden 2048 processed in two 1024-halves via a 32KB bf16 LDS slab.
// Per WMMA: 2x b128 (A) + 2x b128 (B, pre-swizzled) + 1 v_wmma.
__global__ void __launch_bounds__(256, 1)
expert_kernel(const unsigned short* __restrict__ xbf,
              const unsigned int* __restrict__ W1s, const float* __restrict__ b1,
              const unsigned int* __restrict__ W2s, const float* __restrict__ b2,
              const float* __restrict__ gates, const int* __restrict__ cursor,
              const int* __restrict__ lists, float* __restrict__ out, int N) {
  __shared__ __align__(16) unsigned short lds_h[16 * H_DIM];  // 32 KB
  __shared__ int   token_s[16];
  __shared__ float gate_s[16];

  const int e = blockIdx.z;
  const int cnt = cursor[e];
  const int tile = blockIdx.y;
  if (tile * 16 >= cnt) return;

  const int tid  = threadIdx.x;
  const int wave = tid >> 5;
  const int lane = tid & 31;

  if (tid < 16) {
    int idx = tile * 16 + tid;
    if (idx < cnt) {
      int t = lists[e * N + idx];
      token_s[tid] = t;
      gate_s[tid]  = gates[(size_t)t * E_NUM + e];
    } else {                                   // pad partial tile with a valid row, gate 0
      token_s[tid] = lists[e * N + tile * 16];
      gate_s[tid]  = 0.0f;
    }
  }

  const uint4* w1f = (const uint4*)(W1s + (size_t)e * W1_UINTS);  // 64 uint4 per frag
  const uint4* w2f = (const uint4*)(W2s + (size_t)e * W2_UINTS);
  const float* b1e = b1 + (size_t)e * H2_DIM;
  const float* b2e = b2 + (size_t)e * H_DIM;

  const int n     = lane & 15;             // D/B column within 16-tile
  const int mbase = (lane >> 4) << 3;      // D rows: 0..7 or 8..15
  const int kbu4  = (lane >> 4);           // A K-half offset in uint4 units (0 or 1)
  const int lane2 = lane * 2;              // lane offset into a B fragment (uint4 units)

  const v8f vzero = {0.f, 0.f, 0.f, 0.f, 0.f, 0.f, 0.f, 0.f};
  v8f acc[8];
#pragma unroll
  for (int t = 0; t < 8; ++t) acc[t] = vzero;

  for (int kh = 0; kh < 2; ++kh) {
    const int hb = kh * 1024;              // hidden-half base in [0,2048)
    __syncthreads();                       // lds_h free for reuse / token_s visible

    // ---- GEMM1: lds_h[:, wave*128 .. +127] = gelu(x @ W1[:, hb+..] + b1)
    const uint4* xrow4 = (const uint4*)xbf + (size_t)token_s[lane & 15] * (H_DIM / 8);
#pragma unroll 1
    for (int nt = 0; nt < 8; ++nt) {
      const int n0  = wave * 128 + nt * 16;                 // local col base, this half
      const int fb  = ((hb + n0) >> 4) * (H_DIM / 32);      // fragment base (K-major)
      const uint4* bp = w1f + (size_t)fb * 64 + lane2;
      v8f c = vzero;
#pragma unroll 2
      for (int k0 = 0; k0 < H_DIM; k0 += 32) {
        v16bf a = load_a_frag128(xrow4, (k0 >> 3) + kbu4);
        v16bf b = load_b_frag128(bp);
        bp += 64;                                           // next K fragment (+1KB)
        __builtin_prefetch(bp, 0, 1);
        c = __builtin_amdgcn_wmma_f32_16x16x32_bf16(false, a, false, b,
                                                    (short)0, c, false, false);
      }
      const float bias = b1e[hb + n0 + n];
#pragma unroll
      for (int r = 0; r < 8; ++r) {
        int m = mbase + r;
        lds_h[m * H_DIM + n0 + n] = f2bf(gelu_exact(c[r] + bias));
      }
    }
    __syncthreads();                       // h-slab complete

    // ---- GEMM2 partial: acc += h_half @ W2[hb:hb+1024, wave*128 .. +127]
    const uint4* hrow4 = (const uint4*)lds_h + (lane & 15) * (H_DIM / 8);
#pragma unroll 1
    for (int nt = 0; nt < 8; ++nt) {
      const int n0  = wave * 128 + nt * 16;                 // output col base
      const int fb  = (n0 >> 4) * (H2_DIM / 32) + (hb >> 5);
      const uint4* bp = w2f + (size_t)fb * 64 + lane2;
      v8f c = acc[nt];
#pragma unroll 2
      for (int k0 = 0; k0 < H_DIM; k0 += 32) {
        v16bf a = load_a_frag128(hrow4, (k0 >> 3) + kbu4);
        v16bf b = load_b_frag128(bp);
        bp += 64;
        __builtin_prefetch(bp, 0, 1);
        c = __builtin_amdgcn_wmma_f32_16x16x32_bf16(false, a, false, b,
                                                    (short)0, c, false, false);
      }
      acc[nt] = c;
    }
  }

  // ---- epilogue: out[token, :] += gate * (acc + b2); 2 experts/token -> f32 atomics
#pragma unroll 1
  for (int nt = 0; nt < 8; ++nt) {
    const int ncol = wave * 128 + nt * 16 + n;
    const float bias = b2e[ncol];
#pragma unroll
    for (int r = 0; r < 8; ++r) {
      int m = mbase + r;
      float v = gate_s[m] * (acc[nt][r] + bias);
      atomicAdd(&out[(size_t)token_s[m] * H_DIM + ncol], v);
    }
  }
}

// ---------------------------------------------------------------- host launch
extern "C" void kernel_launch(void* const* d_in, const int* in_sizes, int n_in,
                              void* d_out, int out_size, void* d_ws, size_t ws_size,
                              hipStream_t stream) {
  (void)n_in; (void)ws_size;
  const float* x  = (const float*)d_in[0];
  const float* Wr = (const float*)d_in[1];
  const float* W1 = (const float*)d_in[2];
  const float* b1 = (const float*)d_in[3];
  const float* W2 = (const float*)d_in[4];
  const float* b2 = (const float*)d_in[5];
  float* out = (float*)d_out;

  const int N = in_sizes[0] / H_DIM;   // 8192 tokens

  // workspace layout (~81 MB): gates | counts | cursors | lists | xbf | W1s | W2s
  char* ws = (char*)d_ws;
  float* gates = (float*)ws;                 ws += (size_t)N * E_NUM * sizeof(float);
  float* counts = (float*)ws;                ws += 64;
  int* cursor = (int*)ws;                    ws += 64;
  int* lists = (int*)ws;                     ws += (size_t)E_NUM * N * sizeof(int);
  unsigned short* xbf = (unsigned short*)ws; ws += (size_t)N * H_DIM * sizeof(unsigned short);
  unsigned int* W1s = (unsigned int*)ws;     ws += E_NUM * W1_UINTS * sizeof(unsigned int);
  unsigned int* W2s = (unsigned int*)ws;

  init_kernel<<<1024, 256, 0, stream>>>(out, N * H_DIM, counts, cursor);

  // pack: one wave per fragment; E*(4096+4096) fragments
  const int pack_waves = E_NUM * (W1_FRAGS + W2_FRAGS);
  pack_weights_kernel<<<(pack_waves * 32 + 255) / 256, 256, 0, stream>>>(W1, W2, W1s, W2s);

  router_kernel<<<N / 8, 256, 0, stream>>>(x, Wr, gates, counts, cursor, lists, xbf, N);
  aux_kernel<<<1, 32, 0, stream>>>(counts, out, out_size - 1);
  expert_kernel<<<dim3(1, N / 16, E_NUM), 256, 0, stream>>>(
      xbf, W1s, b1, W2s, b2, gates, cursor, lists, out, N);
}
